// VAEModel_7816840479366
// MI455X (gfx1250) — compile-verified, matrix operand order checked by reading, not run on hardware
//
#include <hip/hip_runtime.h>
#include <hip/hip_bf16.h>
#include <math.h>

// ---------------- model constants ----------------
#define BB   8
#define TT   2048
#define BT   (BB*TT)           // 16384 tokens
#define DIM  512
#define KCODES 1024
#define NUM_Q 8
#define TEMPI 10.0f            // 1/TEMP, TEMP = 0.1

typedef __attribute__((ext_vector_type(16))) _Float16 v16h;
typedef __attribute__((ext_vector_type(8)))  float    v8f;

// ---------------- small helpers ----------------
__device__ __forceinline__ float gelu_exact(float x) {
    return 0.5f * x * (1.0f + erff(x * 0.70710678118654752f));
}

__device__ __forceinline__ float gumbel_hash(unsigned s, unsigned r, unsigned n) {
    unsigned h = (s + 1u) * 0x9E3779B9u;
    h ^= r * 0x85EBCA6Bu;  h ^= n * 0xC2B2AE35u;
    h ^= h >> 16; h *= 0x7FEB352Du;
    h ^= h >> 15; h *= 0x846CA68Bu;
    h ^= h >> 16;
    float u = (float)(h & 0x00FFFFFFu) * (1.0f / 16777216.0f) + 1.0e-7f;
    return -__logf(-__logf(u));
}

// ---------------- utility kernels ----------------
__global__ void zero_kernel(float* __restrict__ p, long n) {
    long i = (long)blockIdx.x * blockDim.x + threadIdx.x;
    long stride = (long)gridDim.x * blockDim.x;
    for (; i < n; i += stride) p[i] = 0.0f;
}

// plain fp32 -> f16 copy (row-major), used for the codebook
__global__ void pack_f16_kernel(const float* __restrict__ src, _Float16* __restrict__ dst, long n) {
    long i = (long)blockIdx.x * blockDim.x + threadIdx.x;
    long stride = (long)gridDim.x * blockDim.x;
    for (; i < n; i += stride) dst[i] = (_Float16)src[i];
}

// repack conv weights w[n][ci][d] (fp32) -> wp[n][d*KPAD + ci] (f16), zero padded
template<int CIN, int KPAD, int COUT, int COUTP>
__global__ void pack_w_kernel(const float* __restrict__ w, _Float16* __restrict__ wp) {
    const int total = COUTP * 3 * KPAD;
    int idx = blockIdx.x * blockDim.x + threadIdx.x;
    if (idx >= total) return;
    const int n   = idx / (3 * KPAD);
    const int rem = idx - n * 3 * KPAD;
    const int d   = rem / KPAD;
    const int ci  = rem - d * KPAD;
    float v = 0.0f;
    if (n < COUT && ci < CIN) v = w[((long)n * CIN + ci) * 3 + d];
    wp[idx] = (_Float16)v;
}

__global__ void csq_kernel(const float* __restrict__ cb, float* __restrict__ csq) {
    int k = blockIdx.x * blockDim.x + threadIdx.x;
    if (k >= KCODES) return;
    const float* row = cb + (long)k * DIM;
    float s = 0.0f;
    #pragma unroll 4
    for (int j = 0; j < DIM; ++j) s += row[j] * row[j];
    csq[k] = s;
}

__global__ void mse_kernel(const float* __restrict__ x, const float* __restrict__ y,
                           float* __restrict__ accum, long n) {
    long i = (long)blockIdx.x * blockDim.x + threadIdx.x;
    long stride = (long)gridDim.x * blockDim.x;
    float s = 0.0f;
    for (; i < n; i += stride) { float d = x[i] - y[i]; s += d * d; }
    #pragma unroll
    for (int m = 16; m >= 1; m >>= 1) s += __shfl_xor(s, m, 32);
    if ((threadIdx.x & 31) == 0) atomicAdd(&accum[1], s);
}

__global__ void final_kernel(const float* __restrict__ accum, float* __restrict__ out) {
    float mse    = accum[1] / ((float)BT * 100.0f);
    float commit = accum[0] / ((float)NUM_Q * (float)BT * (float)DIM);
    out[0] = mse + commit;
}

// ---------------- WMMA conv1d (k=3, pad=1, same) + exact GELU ----------------
// im2col GEMM, K = 3*KPAD. B comes from the pre-packed f16 weight plane
// wp[COUTP][3*KPAD] -> one aligned 32-byte vector load per fragment, no guards.
template<int CIN, int KPAD, int COUT, int COUTP>
__global__ __launch_bounds__(128)
void conv1d_gelu_wmma(const float* __restrict__ in,       // [BT, CIN]
                      const _Float16* __restrict__ wp,    // [COUTP, 3*KPAD] packed
                      const float* __restrict__ bias,     // [COUT]
                      float* __restrict__ out)            // [BT, COUT]
{
    constexpr int KCHUNKS = (3 * KPAD) / 32;
    const int lane  = threadIdx.x & 31;
    const int wave  = threadIdx.x >> 5;
    const int nTile = blockIdx.y * 4 + wave;
    const int rowBase = blockIdx.x * 16;
    const int b  = rowBase / TT;
    const int t0 = rowBase - b * TT;

    const int half = lane >> 4;     // K half select
    const int mrow = lane & 15;     // token row (A) / N col (B, C/D)
    const int n    = nTile * 16 + mrow;   // < COUTP always

    v8f acc = {};
    for (int kc = 0; kc < KCHUNKS; ++kc) {
        const int kk0 = kc * 32 + half * 16;
        const int d   = kk0 / KPAD;       // tap 0..2
        const int ci0 = kk0 - d * KPAD;   // channel base within tap

        // ---- A fragment: token row = mrow, im2col K slice (fp32 -> f16) ----
        const int t = t0 + mrow + d - 1;
        const bool tok = (t >= 0) && (t < TT);
        const float* asrc = in + ((long)(b * TT + t)) * CIN + ci0;
        v16h afrag;
        #pragma unroll
        for (int j = 0; j < 16; ++j) {
            float v = 0.0f;
            if (tok && (ci0 + j) < CIN) v = asrc[j];
            afrag[j] = (_Float16)v;
        }

        // ---- B fragment: direct packed-f16 vector load ----
        v16h bfrag = *(const v16h*)(wp + (long)n * (3 * KPAD) + kk0);

        acc = __builtin_amdgcn_wmma_f32_16x16x32_f16(
            false, afrag, false, bfrag, (short)0, acc, false, false);
    }

    // ---- bias + exact GELU + store (C/D layout: VGPR v -> M = v + 8*half, N = mrow) ----
    if (n < COUT) {
        const float bval = bias[n];
        #pragma unroll
        for (int v = 0; v < 8; ++v) {
            const int r = v + 8 * half;
            out[(long)(rowBase + r) * COUT + n] = gelu_exact(acc[v] + bval);
        }
    }
}

// ---------------- one residual-VQ stage (fused GEMM + gumbel argmax + update) ----------------
// One wave owns 16 token rows. A (16x512, f16) resident in VGPRs across all 64
// codebook N-tiles; B loaded straight from pre-packed f16 codebook (no cvt).
__global__ __launch_bounds__(32)
void vq_stage_kernel(float* __restrict__ residual,            // [BT, DIM] in/out
                     float* __restrict__ qout,                // [BT, DIM] accumulated
                     const float* __restrict__ codebook,      // [KCODES, DIM] fp32
                     const _Float16* __restrict__ cbh,        // [KCODES, DIM] f16
                     const float* __restrict__ csq,           // [KCODES]
                     float* __restrict__ accum,               // [0] = commit sum
                     int stage)
{
    const int lane = threadIdx.x;
    const int half = lane >> 4;
    const int mrow = lane & 15;
    const int rowBase = blockIdx.x * 16;

    // Load resident A fragments: 16 K-chunks of 32 (fp32 -> f16, once per stage)
    v16h afrag[16];
    const float* arow = residual + (long)(rowBase + mrow) * DIM;
    #pragma unroll
    for (int kc = 0; kc < 16; ++kc) {
        const float4* s4 = (const float4*)(arow + kc * 32 + half * 16);
        #pragma unroll
        for (int q = 0; q < 4; ++q) {
            float4 f = s4[q];
            afrag[kc][q * 4 + 0] = (_Float16)f.x;
            afrag[kc][q * 4 + 1] = (_Float16)f.y;
            afrag[kc][q * 4 + 2] = (_Float16)f.z;
            afrag[kc][q * 4 + 3] = (_Float16)f.w;
        }
    }

    float bestVal[8];
    int   bestIdx[8];
    #pragma unroll
    for (int v = 0; v < 8; ++v) { bestVal[v] = -3.4e38f; bestIdx[v] = 0; }

    for (int nc = 0; nc < KCODES / 16; ++nc) {
        v8f acc = {};
        const int n = nc * 16 + mrow;
        const _Float16* brow = cbh + (long)n * DIM + half * 16;
        #pragma unroll
        for (int kc = 0; kc < 16; ++kc) {
            v16h bfrag = *(const v16h*)(brow + kc * 32);   // 2x b128, f16 direct
            acc = __builtin_amdgcn_wmma_f32_16x16x32_f16(
                false, afrag[kc], false, bfrag, (short)0, acc, false, false);
        }
        const float cs = csq[n];
        #pragma unroll
        for (int v = 0; v < 8; ++v) {
            const int rloc = v + 8 * half;
            float score = (2.0f * acc[v] - cs) * TEMPI
                        + gumbel_hash((unsigned)stage, (unsigned)(rowBase + rloc), (unsigned)n);
            if (score > bestVal[v]) { bestVal[v] = score; bestIdx[v] = n; }
        }
    }

    // reduce argmax across the 16 lanes of each half (N dimension)
    #pragma unroll
    for (int v = 0; v < 8; ++v) {
        #pragma unroll
        for (int m = 8; m >= 1; m >>= 1) {
            float ov = __shfl_xor(bestVal[v], m, 32);
            int   oi = __shfl_xor(bestIdx[v], m, 32);
            if (ov > bestVal[v]) { bestVal[v] = ov; bestIdx[v] = oi; }
        }
    }
    __shared__ int idx16[16];
    if (mrow == 0) {
        #pragma unroll
        for (int v = 0; v < 8; ++v) idx16[v + 8 * half] = bestIdx[v];
    }
    __syncthreads();

    // update: residual -= q; qout += q; commit += (q - residual_old)^2   (fp32 path)
    float lsum = 0.0f;
    for (int r = 0; r < 16; ++r) {
        const int k = idx16[r];
        const float* qrow = codebook + (long)k * DIM;
        __builtin_prefetch(qrow, 0, 1);   // global_prefetch hint
        float* rrow = residual + (long)(rowBase + r) * DIM;
        float* orow = qout     + (long)(rowBase + r) * DIM;
        for (int j = lane; j < DIM; j += 32) {
            const float qv = qrow[j];
            const float rv = rrow[j];
            const float d  = qv - rv;
            lsum += d * d;
            rrow[j] = rv - qv;
            orow[j] += qv;
        }
    }
    #pragma unroll
    for (int m = 16; m >= 1; m >>= 1) lsum += __shfl_xor(lsum, m, 32);
    if (lane == 0) atomicAdd(&accum[0], lsum);
}

// ---------------- host-side orchestration ----------------
extern "C" void kernel_launch(void* const* d_in, const int* in_sizes, int n_in,
                              void* d_out, int out_size, void* d_ws, size_t ws_size,
                              hipStream_t stream) {
    const float* x        = (const float*)d_in[0];
    const float* enc_w1   = (const float*)d_in[1];
    const float* enc_b1   = (const float*)d_in[2];
    const float* enc_w2   = (const float*)d_in[3];
    const float* enc_b2   = (const float*)d_in[4];
    const float* codebook = (const float*)d_in[5];
    const float* dec_w1   = (const float*)d_in[6];
    const float* dec_b1   = (const float*)d_in[7];
    const float* dec_w2   = (const float*)d_in[8];
    const float* dec_b2   = (const float*)d_in[9];
    float* out = (float*)d_out;

    // workspace layout (fp32 region first, then f16 region)
    float* ws       = (float*)d_ws;
    float* h1       = ws;                        // [BT, 256]
    float* residual = h1 + (long)BT * 256;       // [BT, 512]
    float* qout     = residual + (long)BT * 512; // [BT, 512]
    float* dh       = qout + (long)BT * 512;     // [BT, 256]
    float* y        = dh + (long)BT * 256;       // [BT, 100]
    float* csq      = y + (long)BT * 100;        // [1024]
    float* accum    = csq + KCODES;              // [2] = {commit_sum, mse_sum}

    _Float16* cbh = (_Float16*)(accum + 16);     // [1024, 512] f16 codebook
    _Float16* wp1 = cbh + (long)KCODES * DIM;    // [256, 3*128]
    _Float16* wp2 = wp1 + 256 * 3 * 128;         // [512, 3*256]
    _Float16* wp3 = wp2 + 512 * 3 * 256;         // [256, 3*512]
    _Float16* wp4 = wp3 + 256 * 3 * 512;         // [128, 3*256]

    // init + packing (every call; no cached state)
    zero_kernel<<<512, 256, 0, stream>>>(qout, (long)BT * 512);
    zero_kernel<<<1, 32, 0, stream>>>(accum, 2);
    csq_kernel<<<KCODES / 256, 256, 0, stream>>>(codebook, csq);
    pack_f16_kernel<<<1024, 256, 0, stream>>>(codebook, cbh, (long)KCODES * DIM);
    pack_w_kernel<100, 128, 256, 256><<<(256 * 3 * 128 + 255) / 256, 256, 0, stream>>>(enc_w1, wp1);
    pack_w_kernel<256, 256, 512, 512><<<(512 * 3 * 256 + 255) / 256, 256, 0, stream>>>(enc_w2, wp2);
    pack_w_kernel<512, 512, 256, 256><<<(256 * 3 * 512 + 255) / 256, 256, 0, stream>>>(dec_w1, wp3);
    pack_w_kernel<256, 256, 100, 128><<<(128 * 3 * 256 + 255) / 256, 256, 0, stream>>>(dec_w2, wp4);

    // encoder
    conv1d_gelu_wmma<100, 128, 256, 256><<<dim3(BT / 16, 4), 128, 0, stream>>>(x,  wp1, enc_b1, h1);
    conv1d_gelu_wmma<256, 256, 512, 512><<<dim3(BT / 16, 8), 128, 0, stream>>>(h1, wp2, enc_b2, residual);

    // residual VQ, 8 sequential stages
    for (int s = 0; s < NUM_Q; ++s)
        vq_stage_kernel<<<BT / 16, 32, 0, stream>>>(residual, qout, codebook, cbh, csq, accum, s);

    // decoder
    conv1d_gelu_wmma<512, 512, 256, 256><<<dim3(BT / 16, 4), 128, 0, stream>>>(qout, wp3, dec_b1, dh);
    conv1d_gelu_wmma<256, 256, 100, 128><<<dim3(BT / 16, 2), 128, 0, stream>>>(dh,   wp4, dec_b2, y);

    // losses
    mse_kernel<<<512, 256, 0, stream>>>(x, y, accum, (long)BT * 100);
    final_kernel<<<1, 1, 0, stream>>>(accum, out);
}
